// MyLSTMCell_46540265620152
// MI455X (gfx1250) — matrix operand order repaired and checked by reading, not compiled, not run on hardware
//
#include <hip/hip_runtime.h>

// ---------------------------------------------------------------------------
// LSTM cell for MI455X (gfx1250): bf16 WMMA GEMM fused with gate epilogue.
// B=4096, I=H=1024, G=4.  pre = x*Wx^T + h*Wh^T + bx + bh, then gates.
// Double-buffered async-LDS pipeline: tile k+1 DMA overlaps tile k WMMAs.
// ---------------------------------------------------------------------------

typedef __attribute__((ext_vector_type(16))) __bf16 v16bf;
typedef __attribute__((ext_vector_type(8)))  float  v8f;
typedef int v4i __attribute__((vector_size(16)));   // matches builtin param type

#define NB   4096         // batch
#define NK   1024         // I == H
#define NEL  4194304      // 4096*1024 == 4*1024*1024 (every array is 4M elems)

#define BM   128          // batch rows per block
#define BN   64           // H columns per block
#define KC   64           // K tile staged in LDS
#define LDK  72           // padded K stride (bf16 elems): 144B -> conflict-free

#define A_ELEMS (BM * LDK)        // 9216
#define B_ELEMS (4 * BN * LDK)    // 18432
#define BUF_ELEMS (A_ELEMS + B_ELEMS)         // 27648
#define SMEM_BYTES (2 * BUF_ELEMS * 2)        // 110592 B (double buffer)

// ---- fp32 -> bf16 (RNE) -----------------------------------------------------
__device__ __forceinline__ unsigned short f2bf(float f) {
  union { float f; unsigned u; } v; v.f = f;
  unsigned r = v.u + 0x7FFFu + ((v.u >> 16) & 1u);
  return (unsigned short)(r >> 16);
}
__device__ __forceinline__ void st4bf(unsigned short* d, float4 v) {
  ushort4 o; o.x = f2bf(v.x); o.y = f2bf(v.y); o.z = f2bf(v.z); o.w = f2bf(v.w);
  *(ushort4*)d = o;
}

// ---- async global->LDS copy (CDNA5 path, guarded) ---------------------------
#if defined(__AMDGCN__) && __has_builtin(__builtin_amdgcn_global_load_async_to_lds_b128)
#define ASYNC_CP16(dst, src)                                                   \
  __builtin_amdgcn_global_load_async_to_lds_b128(                              \
      (__attribute__((address_space(1))) v4i*)(unsigned long long)(src),       \
      (__attribute__((address_space(3))) v4i*)(unsigned int)(unsigned long long)(dst), \
      0, 0)
#define HAVE_ASYNC_LDS 1
#else
#define ASYNC_CP16(dst, src) (*(uint4*)(dst) = *(const uint4*)(src))
#define HAVE_ASYNC_LDS 0
#endif

template <int N>
__device__ __forceinline__ void wait_async_lds() {
#if HAVE_ASYNC_LDS
#if __has_builtin(__builtin_amdgcn_s_wait_asynccnt)
  __builtin_amdgcn_s_wait_asynccnt(N);
#elif defined(__AMDGCN__)
  if constexpr (N == 0)
    asm volatile("s_wait_asynccnt 0" ::: "memory");
  else
    asm volatile("s_wait_asynccnt 12" ::: "memory");
#endif
#endif
}

// ---- activations ------------------------------------------------------------
__device__ __forceinline__ float fast_tanh(float x) {
#if defined(__AMDGCN__) && __has_builtin(__builtin_amdgcn_tanhf)
  return __builtin_amdgcn_tanhf(x);
#elif defined(__AMDGCN__) && __has_builtin(__builtin_amdgcn_tanh_f32)
  return __builtin_amdgcn_tanh_f32(x);
#else
  return tanhf(x);
#endif
}
__device__ __forceinline__ float fast_sig(float x) {
  return 1.0f / (1.0f + __expf(-x));
}

// ---------------------------------------------------------------------------
// Pass 1: convert x, prev_h, Wx, Wh (each 4M fp32) to bf16 in workspace.
// ---------------------------------------------------------------------------
__global__ __launch_bounds__(256) void cvt_bf16_all(
    const float* __restrict__ s0, const float* __restrict__ s1,
    const float* __restrict__ s2, const float* __restrict__ s3,
    unsigned short* __restrict__ d0, unsigned short* __restrict__ d1,
    unsigned short* __restrict__ d2, unsigned short* __restrict__ d3) {
  unsigned i = (blockIdx.x * 256u + threadIdx.x) * 4u;
  if (i >= NEL) return;
  st4bf(d0 + i, *(const float4*)(s0 + i));
  st4bf(d1 + i, *(const float4*)(s1 + i));
  st4bf(d2 + i, *(const float4*)(s2 + i));
  st4bf(d3 + i, *(const float4*)(s3 + i));
}

// ---------------------------------------------------------------------------
// Stage one 128x64 A tile + 4x(64x64) B tiles into LDS (12 b128 ops/thread).
// ---------------------------------------------------------------------------
__device__ __forceinline__ void stage_tile(
    const unsigned short* __restrict__ Ag, const unsigned short* __restrict__ Bg,
    unsigned short* As, unsigned short* Bs, int rowB, int cb, int kb, int t) {
#pragma unroll
  for (int i = 0; i < 4; ++i) {           // A: 1024 segs / 256 threads
    int seg = i * 256 + t;
    int r = seg >> 3, kk = (seg & 7) * 8;
    ASYNC_CP16(As + r * LDK + kk, Ag + (rowB + r) * NK + kb + kk);
  }
#pragma unroll
  for (int i = 0; i < 8; ++i) {           // B: 2048 segs / 256 threads
    int seg = i * 256 + t;
    int g = seg >> 9, cc = (seg >> 3) & 63, kk = (seg & 7) * 8;
    ASYNC_CP16(Bs + (g * BN + cc) * LDK + kk,
               Bg + (g * NK + cb + cc) * NK + kb + kk);
  }
}

// ---------------------------------------------------------------------------
// Pass 2: fused 4-gate GEMM (bf16 WMMA, fp32 acc) + LSTM epilogue.
// Grid: (B/BM, H/BN). 256 threads = 8 waves; wave w owns rows [16w,16w+16)
// of the block tile and all 4 gate accumulators for its 16x64 slice.
// ---------------------------------------------------------------------------
__global__ __launch_bounds__(256) void lstm_wmma(
    const unsigned short* __restrict__ xb,  const unsigned short* __restrict__ hb,
    const unsigned short* __restrict__ wxb, const unsigned short* __restrict__ whb,
    const float* __restrict__ prev_c,
    const float* __restrict__ bx, const float* __restrict__ bh,
    float* __restrict__ out) {
  extern __shared__ __align__(16) unsigned short smem[];
  // buffer b: A tile at smem + b*BUF_ELEMS, B tiles right after A
  // (no pointer arrays: aggregate inits of LDS addrs break the linker)

  const int t    = threadIdx.x;
  const int w    = t >> 5;            // wave id 0..7
  const int lane = t & 31;
  const int nloc = lane & 15;         // N (cols) / M (A rows) within tile
  const int hi16 = lane >> 4;         // lane half selects K-chunk / M-half
  const int rowB = blockIdx.x * BM;   // block batch-row base
  const int cb   = blockIdx.y * BN;   // block H-column base

  v8f acc[4][4];
  const v8f vzero = {0.f, 0.f, 0.f, 0.f, 0.f, 0.f, 0.f, 0.f};
#pragma unroll
  for (int g = 0; g < 4; ++g)
#pragma unroll
    for (int n = 0; n < 4; ++n) acc[g][n] = vzero;

  union Frag { uint4 u4[2]; v16bf v; };

  // ---- software pipeline over 32 flattened K-tiles (2 phases x 16 tiles) ----
  stage_tile(xb, wxb, smem, smem + A_ELEMS, rowB, cb, 0, t);

#pragma unroll 1
  for (int kt = 0; kt < 32; ++kt) {
    const int cur = kt & 1;
    if (kt + 1 < 32) {
      const int nk  = kt + 1;
      const unsigned short* Agn = (nk >> 4) ? hb  : xb;
      const unsigned short* Bgn = (nk >> 4) ? whb : wxb;
      unsigned short* Asn = smem + (nk & 1) * BUF_ELEMS;
      stage_tile(Agn, Bgn, Asn, Asn + A_ELEMS, rowB, cb, (nk & 15) * KC, t);
      wait_async_lds<12>();   // only tile kt's 12 ops must be complete
    } else {
      wait_async_lds<0>();
    }
    __syncthreads();          // tile kt visible to all waves

    const unsigned short* Asc = smem + cur * BUF_ELEMS;
    const unsigned short* Bsc = Asc + A_ELEMS;

    // A fragments for both K-steps (K 0-31 and 32-63 of this tile)
    Frag a0, a1;
    const unsigned short* arow = Asc + (w * 16 + nloc) * LDK;
    a0.u4[0] = *(const uint4*)(arow + hi16 * 8);
    a0.u4[1] = *(const uint4*)(arow + 16 + hi16 * 8);
    a1.u4[0] = *(const uint4*)(arow + 32 + hi16 * 8);
    a1.u4[1] = *(const uint4*)(arow + 48 + hi16 * 8);

#pragma unroll
    for (int g = 0; g < 4; ++g) {
      // group-load 4 B fragments -> one DS wait per 4 WMMAs
      Frag b0[4];
#pragma unroll
      for (int nt = 0; nt < 4; ++nt) {
        const unsigned short* brow = Bsc + (g * BN + nt * 16 + nloc) * LDK;
        b0[nt].u4[0] = *(const uint4*)(brow + hi16 * 8);
        b0[nt].u4[1] = *(const uint4*)(brow + 16 + hi16 * 8);
      }
#pragma unroll
      for (int nt = 0; nt < 4; ++nt)
        acc[g][nt] = __builtin_amdgcn_wmma_f32_16x16x32_bf16(
            false, a0.v, false, b0[nt].v, (short)0, acc[g][nt], false, false);

      Frag b1[4];
#pragma unroll
      for (int nt = 0; nt < 4; ++nt) {
        const unsigned short* brow = Bsc + (g * BN + nt * 16 + nloc) * LDK + 32;
        b1[nt].u4[0] = *(const uint4*)(brow + hi16 * 8);
        b1[nt].u4[1] = *(const uint4*)(brow + 16 + hi16 * 8);
      }
#pragma unroll
      for (int nt = 0; nt < 4; ++nt)
        acc[g][nt] = __builtin_amdgcn_wmma_f32_16x16x32_bf16(
            false, a1.v, false, b1[nt].v, (short)0, acc[g][nt], false, false);
    }
    __syncthreads();          // all waves done reading buf[cur] before reuse
  }

  // --- epilogue: bias + gates, fully in-register (this wave owns g,i,f,o)
#pragma unroll
  for (int nt = 0; nt < 4; ++nt) {
    const int col = cb + nt * 16 + nloc;
    const float b0 = bx[0 * NK + col] + bh[0 * NK + col];
    const float b1 = bx[1 * NK + col] + bh[1 * NK + col];
    const float b2 = bx[2 * NK + col] + bh[2 * NK + col];
    const float b3 = bx[3 * NK + col] + bh[3 * NK + col];
#pragma unroll
    for (int r = 0; r < 8; ++r) {
      const int row = rowB + w * 16 + hi16 * 8 + r;  // C/D layout: M = r (+8)
      const float gg = fast_tanh(acc[0][nt][r] + b0);
      const float ii = fast_sig (acc[1][nt][r] + b1);
      const float ff = fast_sig (acc[2][nt][r] + b2);
      const float oo = fast_sig (acc[3][nt][r] + b3);
      const float cp = prev_c[row * NK + col];
      const float c  = gg * ii + cp * ff;
      const float hv = fast_tanh(c) * oo;
      out[row * NK + col]       = hv;   // h
      out[NEL + row * NK + col] = c;    // c
    }
  }
}

// ---------------------------------------------------------------------------
extern "C" void kernel_launch(void* const* d_in, const int* in_sizes, int n_in,
                              void* d_out, int out_size, void* d_ws, size_t ws_size,
                              hipStream_t stream) {
  (void)in_sizes; (void)n_in; (void)out_size; (void)ws_size;
  const float* x  = (const float*)d_in[0];
  const float* ph = (const float*)d_in[1];
  const float* pc = (const float*)d_in[2];
  const float* Wx = (const float*)d_in[3];
  const float* Wh = (const float*)d_in[4];
  const float* bx = (const float*)d_in[5];
  const float* bh = (const float*)d_in[6];
  float* out = (float*)d_out;

  unsigned short* ws  = (unsigned short*)d_ws;
  unsigned short* xb  = ws;               // 4M bf16
  unsigned short* hb  = ws + NEL;         // 4M
  unsigned short* wxb = ws + 2 * NEL;     // 4M
  unsigned short* whb = ws + 3 * NEL;     // 4M  (32 MB total)

  cvt_bf16_all<<<NEL / (256 * 4), 256, 0, stream>>>(x, ph, Wx, Wh,
                                                    xb, hb, wxb, whb);
  dim3 grid(NB / BM, NK / BN);  // 32 x 16
  lstm_wmma<<<grid, 256, SMEM_BYTES, stream>>>(xb, hb, wxb, whb, pc, bx, bh, out);
}
